// BigBirdRegressor_cat_42150809043589
// MI455X (gfx1250) — compile-verified, matrix operand order checked by reading, not run on hardware
//
#include <hip/hip_runtime.h>

typedef __bf16 v16bf __attribute__((ext_vector_type(16)));
typedef float v8f    __attribute__((ext_vector_type(8)));
typedef unsigned short u16;
typedef unsigned int   u32;

// ---------------------------------------------------------------- helpers ---

__device__ __forceinline__ u16 f2bf(float f) {
    u32 u = __float_as_uint(f);
    u32 r = u + 0x7FFFu + ((u >> 16) & 1u);
    return (u16)(r >> 16);
}

struct Frag { union { v16bf v; u32 u[8]; }; };

// Load one 16x32 bf16 A/B fragment stripe for this lane from a row-major
// (contiguous along K) bf16 buffer. p0 points at (row_or_col, k0).
// ISA layout: lane<16 holds K = {0..7, 16..23}, lane>=16 holds K = {8..15, 24..31}.
__device__ __forceinline__ void load_frag(Frag& f, const u16* p0, int lane) {
    const u16* p = p0 + ((lane >> 4) << 3);
#pragma unroll
    for (int j = 0; j < 4; ++j) {
        f.u[j]     = *(const u32*)(p + 2 * j);
        f.u[j + 4] = *(const u32*)(p + 16 + 2 * j);
    }
}

__device__ __forceinline__ v8f wmma_bf16(const Frag& a, const Frag& b, v8f c) {
    return __builtin_amdgcn_wmma_f32_16x16x32_bf16(false, a.v, false, b.v,
                                                   (short)0, c, false, false);
}

// ------------------------------------------------------------------ embed ---

__global__ void __launch_bounds__(256)
embed_ln_kernel(const int* __restrict__ ids, const float* __restrict__ etok,
                const float* __restrict__ epos, const float* __restrict__ g,
                const float* __restrict__ beta, float* __restrict__ x,
                u16* __restrict__ xbf) {
    int t = blockIdx.x;          // token index, 0..8191
    int s = t & 4095;
    int id = ids[t];
    __shared__ float buf[768];
    __shared__ float red[256];
    int tid = threadIdx.x;
    float local = 0.f;
    for (int d = tid; d < 768; d += 256) {
        float v = etok[id * 768 + d] + epos[s * 768 + d];
        buf[d] = v; local += v;
    }
    red[tid] = local; __syncthreads();
    for (int st = 128; st > 0; st >>= 1) { if (tid < st) red[tid] += red[tid + st]; __syncthreads(); }
    float mean = red[0] * (1.f / 768.f);
    __syncthreads();
    local = 0.f;
    for (int d = tid; d < 768; d += 256) { float dv = buf[d] - mean; local += dv * dv; }
    red[tid] = local; __syncthreads();
    for (int st = 128; st > 0; st >>= 1) { if (tid < st) red[tid] += red[tid + st]; __syncthreads(); }
    float rstd = rsqrtf(red[0] * (1.f / 768.f) + 1e-12f);
    for (int d = tid; d < 768; d += 256) {
        float v = (buf[d] - mean) * rstd * g[d] + beta[d];
        x[t * 768 + d] = v;
        xbf[t * 768 + d] = f2bf(v);
    }
}

// ------------------------------------------------------ residual+layernorm ---

__global__ void __launch_bounds__(256)
residual_ln_kernel(const float* __restrict__ xin, const float* __restrict__ y,
                   const float* __restrict__ g, const float* __restrict__ beta,
                   float* __restrict__ xout, u16* __restrict__ xbf) {
    int t = blockIdx.x;
    __shared__ float buf[768];
    __shared__ float red[256];
    int tid = threadIdx.x;
    float local = 0.f;
    for (int d = tid; d < 768; d += 256) {
        float v = xin[t * 768 + d] + y[t * 768 + d];
        buf[d] = v; local += v;
    }
    red[tid] = local; __syncthreads();
    for (int st = 128; st > 0; st >>= 1) { if (tid < st) red[tid] += red[tid + st]; __syncthreads(); }
    float mean = red[0] * (1.f / 768.f);
    __syncthreads();
    local = 0.f;
    for (int d = tid; d < 768; d += 256) { float dv = buf[d] - mean; local += dv * dv; }
    red[tid] = local; __syncthreads();
    for (int st = 128; st > 0; st >>= 1) { if (tid < st) red[tid] += red[tid + st]; __syncthreads(); }
    float rstd = rsqrtf(red[0] * (1.f / 768.f) + 1e-12f);
    for (int d = tid; d < 768; d += 256) {
        float v = (buf[d] - mean) * rstd * g[d] + beta[d];
        xout[t * 768 + d] = v;
        xbf[t * 768 + d] = f2bf(v);
    }
}

// --------------------------------------------------- weight convert (W->Wt) ---

__global__ void wt_convert_kernel(const float* __restrict__ W, u16* __restrict__ Wt,
                                  int K, int N) {
    int i = blockIdx.x * 256 + threadIdx.x;
    if (i >= K * N) return;
    int k = i / N, n = i % N;
    Wt[n * K + k] = f2bf(W[i]);
}

// ---------------------------------------------------------------- WMMA GEMM ---
// Y[M,N] = A[M,K](bf16) @ Bt[N,K]^T(bf16) + bias ; optional GELU; optional
// f32 and/or bf16 outputs. Register-blocked: each wave computes a 32x64 tile
// (2 A-frags x 4 B-frags -> 8 WMMAs per 32-K step); 8 waves -> 128x128/block.

__global__ void __launch_bounds__(256)
gemm_kernel(const u16* __restrict__ A, const u16* __restrict__ Bt,
            const float* __restrict__ bias, float* __restrict__ Yf,
            u16* __restrict__ Ybf, int M, int N, int K, int act) {
    int lane = threadIdx.x & 31;
    int wv   = threadIdx.x >> 5;
    int mBlocks = M >> 7;
    int bm = (blockIdx.x % mBlocks) << 7;
    int bn = (blockIdx.x / mBlocks) << 7;
    int wm = bm + ((wv & 3) << 5);   // 4 waves along M (32 rows each)
    int wn = bn + ((wv >> 2) << 6);  // 2 waves along N (64 cols each)

    const u16* ar[2];
    ar[0] = A + (long)(wm + (lane & 15)) * K;
    ar[1] = ar[0] + 16 * (long)K;
    const u16* br[4];
    br[0] = Bt + (long)(wn + (lane & 15)) * K;
    br[1] = br[0] + 16 * (long)K;
    br[2] = br[0] + 32 * (long)K;
    br[3] = br[0] + 48 * (long)K;

    v8f zacc = {};
    v8f acc[2][4];
#pragma unroll
    for (int i = 0; i < 2; ++i)
#pragma unroll
        for (int j = 0; j < 4; ++j) acc[i][j] = zacc;

    for (int k0 = 0; k0 < K; k0 += 32) {
        __builtin_prefetch(ar[0] + k0 + 512, 0, 1);   // global_prefetch_b8
        __builtin_prefetch(br[0] + k0 + 512, 0, 1);
        Frag fa[2], fb[4];
#pragma unroll
        for (int i = 0; i < 2; ++i) load_frag(fa[i], ar[i] + k0, lane);
#pragma unroll
        for (int j = 0; j < 4; ++j) load_frag(fb[j], br[j] + k0, lane);
#pragma unroll
        for (int i = 0; i < 2; ++i)
#pragma unroll
            for (int j = 0; j < 4; ++j)
                acc[i][j] = wmma_bf16(fa[i], fb[j], acc[i][j]);
    }

    int mb = (lane >> 4) << 3;
#pragma unroll
    for (int i = 0; i < 2; ++i) {
#pragma unroll
        for (int j = 0; j < 4; ++j) {
            int col = wn + j * 16 + (lane & 15);
            float bv = bias[col];
#pragma unroll
            for (int r = 0; r < 8; ++r) {
                float v = acc[i][j][r] + bv;
                if (act) {  // tanh-approx GELU
                    float u = v;
                    v = 0.5f * u * (1.f + tanhf(0.7978845608f * (u + 0.044715f * u * u * u)));
                }
                long off = (long)(wm + i * 16 + mb + r) * N + col;
                if (Yf)  Yf[off]  = v;
                if (Ybf) Ybf[off] = f2bf(v);
            }
        }
    }
}

// ------------------------------------------------------------- reshapes ------
// Y[B*S, D] f32 -> q/k layout [B,H,n,BLK,DH] bf16, and V -> [B,H,n,DH,BLK].

__global__ void reshape_qk_kernel(const float* __restrict__ Y, u16* __restrict__ out) {
    int i = blockIdx.x * 256 + threadIdx.x;
    if (i >= 8192 * 768) return;
    int d = i & 63;
    int h = (i >> 6) % 12;
    int s = (i / 768) & 4095;
    int b = i / (768 * 4096);
    out[(((b * 12 + h) * 64 + (s >> 6)) * 64 + (s & 63)) * 64 + d] = f2bf(Y[i]);
}

__global__ void reshape_vt_kernel(const float* __restrict__ Y, u16* __restrict__ out) {
    int i = blockIdx.x * 256 + threadIdx.x;
    if (i >= 8192 * 768) return;
    int d = i & 63;
    int h = (i >> 6) % 12;
    int s = (i / 768) & 4095;
    int b = i / (768 * 4096);
    out[(((b * 12 + h) * 64 + (s >> 6)) * 64 + d) * 64 + (s & 63)] = f2bf(Y[i]);
}

// ---------------------------------------------------- block-sparse attention ---
// One block per (b,h,qblock,qtile16). Streaming softmax over key chunks of 512:
// sparse blocks -> 1 chunk of gathered {global,window,random} blocks,
// first/last blocks -> 8 chunks covering all 4096 keys (full attention rows).

__global__ void __launch_bounds__(256)
attn_kernel(const u16* __restrict__ qbf, const u16* __restrict__ kbf,
            const u16* __restrict__ vtbf, const int* __restrict__ rand_blocks,
            u16* __restrict__ aobf) {
    const int n = 64, H = 12;
    int tile = blockIdx.x;
    int qt = tile & 3;
    int qb = (tile >> 2) & 63;
    int bh = tile >> 8;          // b*H + h
    int h  = bh % H;
    int b  = bh / H;
    int tid = threadIdx.x, lane = tid & 31, wv = tid >> 5;

    __shared__ float sc[16][512];    // score chunk (f32)
    __shared__ u16   pb[16][512];    // prob chunk (bf16)
    __shared__ float ctxs[16][64];   // f32 context accumulator
    __shared__ float mrow[16], lrow[16];
    __shared__ int   list[64];

    bool full = (qb == 0) || (qb == n - 1);
    int nchunks = full ? 8 : 1;
    if (full) {
        for (int i = tid; i < 64; i += 256) list[i] = i;
    } else if (tid == 0) {
        list[0] = 0;
        list[1] = qb - 1 > 0 ? qb - 1 : 0;
        list[2] = qb;
        list[3] = qb + 1 < n - 1 ? qb + 1 : n - 1;
        list[4] = n - 1;
        const int* rb = rand_blocks + (h * n + qb) * 3;
        list[5] = rb[0]; list[6] = rb[1]; list[7] = rb[2];
    }
    for (int i = tid; i < 16 * 64; i += 256) ctxs[i >> 6][i & 63] = 0.f;
    if (tid < 16) { mrow[tid] = -3.0e38f; lrow[tid] = 0.f; }
    __syncthreads();

    // Q fragments for this 16-row tile (2 k-steps over DH=64), loaded once.
    const u16* qbase = qbf + (((long)(bh) * 64 + qb) * 64 + qt * 16) * 64;
    Frag afq[2];
#pragma unroll
    for (int ks = 0; ks < 2; ++ks)
        load_frag(afq[ks], qbase + (long)(lane & 15) * 64 + ks * 32, lane);
    const float scale = 0.125f;  // 1/sqrt(64)

    for (int c = 0; c < nchunks; ++c) {
        // -- phase 1: scores for 512 keys; wave wv owns key-tiles wv*4..wv*4+3
#pragma unroll
        for (int t = 0; t < 4; ++t) {
            int kt = wv * 4 + t;
            int gkey = c * 512 + kt * 16 + (lane & 15);
            int blk = list[gkey >> 6];
            int krow = gkey & 63;
            const u16* kb = kbf + (((long)(bh) * 64 + blk) * 64 + krow) * 64;
            v8f acc = {};
#pragma unroll
            for (int ks = 0; ks < 2; ++ks) {
                Frag fb;
                load_frag(fb, kb + ks * 32, lane);
                acc = wmma_bf16(afq[ks], fb, acc);
            }
            int mb = (lane >> 4) << 3;
            int col = kt * 16 + (lane & 15);
#pragma unroll
            for (int r = 0; r < 8; ++r) sc[mb + r][col] = acc[r] * scale;
        }
        __syncthreads();

        // -- phase 2: online softmax update. Wave wv: rows wv*2 + (lane>>4).
        // Lane sl covers columns {sl, sl+16, ...} -> conflict-free LDS banks.
        {
            int r  = wv * 2 + (lane >> 4);
            int sl = lane & 15;
            float mloc = -3.0e38f;
            for (int cb = 0; cb < 32; ++cb) mloc = fmaxf(mloc, sc[r][cb * 16 + sl]);
            for (int m = 1; m < 16; m <<= 1) mloc = fmaxf(mloc, __shfl_xor(mloc, m, 32));
            float mold = mrow[r];
            float mnew = fmaxf(mold, mloc);
            float ssum = 0.f;
            for (int cb = 0; cb < 32; ++cb) {
                float e = __expf(sc[r][cb * 16 + sl] - mnew);
                pb[r][cb * 16 + sl] = f2bf(e);
                ssum += e;
            }
            for (int m = 1; m < 16; m <<= 1) ssum += __shfl_xor(ssum, m, 32);
            float alpha = __expf(mold - mnew);
            if (sl == 0) { mrow[r] = mnew; lrow[r] = lrow[r] * alpha + ssum; }
            for (int d = sl; d < 64; d += 16) ctxs[r][d] *= alpha;
        }
        __syncthreads();

        // -- phase 3: ctx += P(16x512, bf16 in LDS) @ V(512x64).
        // All 8 waves: 4 n-tiles x 2 K-halves; partials merged via LDS atomics.
        {
            int nt    = wv & 3;
            int kbeg  = (wv >> 2) << 3;   // 0 or 8
            v8f acc = {};
            for (int ks = kbeg; ks < kbeg + 8; ++ks) {
                Frag pa;
                {
                    const u16* p = &pb[lane & 15][ks * 32 + ((lane >> 4) << 3)];
#pragma unroll
                    for (int j = 0; j < 4; ++j) {
                        pa.u[j]     = *(const u32*)(p + 2 * j);
                        pa.u[j + 4] = *(const u32*)(p + 16 + 2 * j);
                    }
                }
                int g0 = c * 512 + ks * 32;
                int blk = list[g0 >> 6];
                int r0 = g0 & 63;
                const u16* vb = vtbf + (((long)(bh) * 64 + blk) * 64 + nt * 16 + (lane & 15)) * 64 + r0;
                Frag fb;
                load_frag(fb, vb, lane);
                acc = wmma_bf16(pa, fb, acc);
            }
            int mb = (lane >> 4) << 3;
            int col = nt * 16 + (lane & 15);
#pragma unroll
            for (int r = 0; r < 8; ++r) atomicAdd(&ctxs[mb + r][col], acc[r]);
        }
        __syncthreads();
    }

    // -- output: normalize and scatter to [B,S,D] bf16 (head-major columns)
    for (int i = tid; i < 16 * 64; i += 256) {
        int r = i >> 6, d = i & 63;
        float v = ctxs[r][d] / lrow[r];
        long srow = (long)b * 4096 + qb * 64 + qt * 16 + r;
        aobf[srow * 768 + h * 64 + d] = f2bf(v);
    }
}

// ------------------------------------------------------------- pooling/head ---

__global__ void zero_kernel(float* __restrict__ p, int n) {
    int i = blockIdx.x * 256 + threadIdx.x;
    if (i < n) p[i] = 0.f;
}

// Grid: B*8 blocks; block (b, chunk) accumulates 512 tokens into pooled[b].
__global__ void __launch_bounds__(256)
pool_kernel(const float* __restrict__ x, float* __restrict__ pooled) {
    int b = blockIdx.x >> 3;
    int c = blockIdx.x & 7;
    int tid = threadIdx.x;
    float s0 = 0.f, s1 = 0.f, s2 = 0.f;
    for (int t = c * 512; t < c * 512 + 512; ++t) {
        const float* row = x + ((long)b * 4096 + t) * 768;
        s0 += row[tid];
        s1 += row[tid + 256];
        s2 += row[tid + 512];
    }
    atomicAdd(&pooled[b * 768 + tid],       s0 * (1.f / 4096.f));
    atomicAdd(&pooled[b * 768 + tid + 256], s1 * (1.f / 4096.f));
    atomicAdd(&pooled[b * 768 + tid + 512], s2 * (1.f / 4096.f));
}

__global__ void __launch_bounds__(256)
head_kernel(const float* __restrict__ pooled, const float* __restrict__ fcw,
            const float* __restrict__ fcb, float* __restrict__ out) {
    int b = blockIdx.x;
    __shared__ float red[256];
    int tid = threadIdx.x;
    float s = 0.f;
    for (int d = tid; d < 768; d += 256) s += pooled[b * 768 + d] * fcw[d];
    red[tid] = s; __syncthreads();
    for (int st = 128; st > 0; st >>= 1) { if (tid < st) red[tid] += red[tid + st]; __syncthreads(); }
    if (tid == 0) out[b] = red[0] + fcb[0];
}

// ------------------------------------------------------------------ launch ---

extern "C" void kernel_launch(void* const* d_in, const int* in_sizes, int n_in,
                              void* d_out, int out_size, void* d_ws, size_t ws_size,
                              hipStream_t stream) {
    (void)in_sizes; (void)n_in; (void)out_size; (void)ws_size;
    const int*   ids      = (const int*)d_in[0];
    const int*   rand_b   = (const int*)d_in[1];
    const float* emb_tok  = (const float*)d_in[2];
    const float* emb_pos  = (const float*)d_in[3];
    const float* ln_emb_g = (const float*)d_in[4];
    const float* ln_emb_b = (const float*)d_in[5];
    const float* Wq = (const float*)d_in[6];  const float* bq = (const float*)d_in[7];
    const float* Wk = (const float*)d_in[8];  const float* bk = (const float*)d_in[9];
    const float* Wv = (const float*)d_in[10]; const float* bv = (const float*)d_in[11];
    const float* Wo = (const float*)d_in[12]; const float* bo = (const float*)d_in[13];
    const float* ln1_g = (const float*)d_in[14]; const float* ln1_b = (const float*)d_in[15];
    const float* W1 = (const float*)d_in[16]; const float* b1 = (const float*)d_in[17];
    const float* W2 = (const float*)d_in[18]; const float* b2 = (const float*)d_in[19];
    const float* ln2_g = (const float*)d_in[20]; const float* ln2_b = (const float*)d_in[21];
    const float* fc_w = (const float*)d_in[22]; const float* fc_b = (const float*)d_in[23];

    const long M = 8192;        // B*S
    char* p = (char*)d_ws;
    auto carve = [&](size_t sz) { void* r = (void*)p; p += (sz + 255) & ~(size_t)255; return r; };
    float* x      = (float*)carve(M * 768 * 4);
    u16*   xbf    = (u16*)  carve(M * 768 * 2);
    float* y      = (float*)carve(M * 3072 * 4);
    u16*   qbf    = (u16*)  carve(M * 768 * 2);
    u16*   kbf    = (u16*)  carve(M * 768 * 2);
    u16*   vtbf   = (u16*)  carve(M * 768 * 2);
    u16*   h1bf   = (u16*)  carve(M * 3072 * 2);
    u16*   aobf   = (u16*)  carve(M * 768 * 2);
    u16*   wt     = (u16*)  carve((size_t)768 * 3072 * 2);
    float* pooled = (float*)carve(2 * 768 * 4);

    const int elemGrid = (int)((M * 768 + 255) / 256);
    const int gemmGrid768  = (int)(M / 128) * (768 / 128);   // 384
    const int gemmGrid3072 = (int)(M / 128) * (3072 / 128);  // 1536

    embed_ln_kernel<<<8192, 256, 0, stream>>>(ids, emb_tok, emb_pos, ln_emb_g, ln_emb_b, x, xbf);

    for (int l = 0; l < 2; ++l) {
        const float* Wq_l = Wq + (size_t)l * 768 * 768;
        const float* Wk_l = Wk + (size_t)l * 768 * 768;
        const float* Wv_l = Wv + (size_t)l * 768 * 768;
        const float* Wo_l = Wo + (size_t)l * 768 * 768;
        const float* W1_l = W1 + (size_t)l * 768 * 3072;
        const float* W2_l = W2 + (size_t)l * 3072 * 768;

        // Q
        wt_convert_kernel<<<(768 * 768 + 255) / 256, 256, 0, stream>>>(Wq_l, wt, 768, 768);
        gemm_kernel<<<gemmGrid768, 256, 0, stream>>>(xbf, wt, bq + l * 768, y, nullptr, 8192, 768, 768, 0);
        reshape_qk_kernel<<<elemGrid, 256, 0, stream>>>(y, qbf);
        // K
        wt_convert_kernel<<<(768 * 768 + 255) / 256, 256, 0, stream>>>(Wk_l, wt, 768, 768);
        gemm_kernel<<<gemmGrid768, 256, 0, stream>>>(xbf, wt, bk + l * 768, y, nullptr, 8192, 768, 768, 0);
        reshape_qk_kernel<<<elemGrid, 256, 0, stream>>>(y, kbf);
        // V (d-major for WMMA B-fragments of P@V)
        wt_convert_kernel<<<(768 * 768 + 255) / 256, 256, 0, stream>>>(Wv_l, wt, 768, 768);
        gemm_kernel<<<gemmGrid768, 256, 0, stream>>>(xbf, wt, bv + l * 768, y, nullptr, 8192, 768, 768, 0);
        reshape_vt_kernel<<<elemGrid, 256, 0, stream>>>(y, vtbf);

        // block-sparse attention with full first/last rows
        attn_kernel<<<2 * 12 * 64 * 4, 256, 0, stream>>>(qbf, kbf, vtbf, rand_b, aobf);

        // output projection + residual LN1
        wt_convert_kernel<<<(768 * 768 + 255) / 256, 256, 0, stream>>>(Wo_l, wt, 768, 768);
        gemm_kernel<<<gemmGrid768, 256, 0, stream>>>(aobf, wt, bo + l * 768, y, nullptr, 8192, 768, 768, 0);
        residual_ln_kernel<<<8192, 256, 0, stream>>>(x, y, ln1_g + l * 768, ln1_b + l * 768, x, xbf);

        // FFN: GELU(x@W1+b1)@W2+b2, residual LN2
        wt_convert_kernel<<<(768 * 3072 + 255) / 256, 256, 0, stream>>>(W1_l, wt, 768, 3072);
        gemm_kernel<<<gemmGrid3072, 256, 0, stream>>>(xbf, wt, b1 + l * 3072, nullptr, h1bf, 8192, 3072, 768, 1);
        wt_convert_kernel<<<(3072 * 768 + 255) / 256, 256, 0, stream>>>(W2_l, wt, 3072, 768);
        gemm_kernel<<<gemmGrid768, 256, 0, stream>>>(h1bf, wt, b2 + l * 768, y, nullptr, 8192, 768, 3072, 0);
        residual_ln_kernel<<<8192, 256, 0, stream>>>(x, y, ln2_g + l * 768, ln2_b + l * 768, x, xbf);
    }

    zero_kernel<<<6, 256, 0, stream>>>(pooled, 2 * 768);
    pool_kernel<<<16, 256, 0, stream>>>(x, pooled);
    head_kernel<<<2, 256, 0, stream>>>(pooled, fc_w, fc_b, (float*)d_out);
}
